// MultiHeadAttention_48739288875068
// MI455X (gfx1250) — matrix operand order repaired and checked by reading, not compiled, not run on hardware
//
#include <hip/hip_runtime.h>
#include <hip/hip_bf16.h>

// ---------------------------------------------------------------------------
// MHA + LayerNorm for MI455X (gfx1250, wave32, WMMA f16 path + TDM staging)
// ---------------------------------------------------------------------------

typedef _Float16 half_t;
typedef __attribute__((ext_vector_type(16))) _Float16 v16h;
typedef __attribute__((ext_vector_type(8)))  float    v8f;

union frag16 { v16h v; unsigned u[8]; _Float16 h[16]; };

#define BB 4
#define SS 2048
#define DM 1024
#define NH 16
#define DK 64

// ----------------------- Tensor Data Mover (CDNA5) -------------------------
#if defined(__gfx1250__) && __has_builtin(__builtin_amdgcn_tensor_load_to_lds)
#define USE_TDM 1
#else
#define USE_TDM 0
#endif

#if USE_TDM
typedef __attribute__((ext_vector_type(4))) unsigned tdm_g0_t;
typedef __attribute__((ext_vector_type(8))) int      tdm_g1_t;
typedef __attribute__((ext_vector_type(4))) int      tdm_g2_t;
typedef __attribute__((ext_vector_type(8))) int      tdm_g4_t;
typedef __attribute__((address_space(3))) char       lds_char_t;

__device__ __forceinline__ unsigned lds_offset_of(const void* p) {
  return (unsigned)(unsigned long long)(lds_char_t*)p;
}

// 2-D tile DMA: rows x (row_units*8 bytes), global row stride = stride_units*8B.
// pad_w0_bits: pre-assembled pad_enable/pad_interval/pad_amount bits for w0.
__device__ __forceinline__ void tdm_load_2d(
    const void* gaddr, unsigned lds_byte_off,
    unsigned row_units, unsigned rows, unsigned stride_units,
    unsigned tensor_d0_units, unsigned tensor_d1, unsigned pad_w0_bits) {
  unsigned long long ga = (unsigned long long)gaddr;
  tdm_g0_t g0 = {
      1u,                                     // count = 1 (valid user D#)
      lds_byte_off,                           // lds_addr
      (unsigned)ga,                           // global_addr[31:0]
      (unsigned)((ga >> 32) & 0x1FFFFFFu) |   // global_addr[56:32]
          0x80000000u };                      // type = 2 ("image")
  tdm_g1_t g1 = {
      (int)(0x30000u | pad_w0_bits),          // data_size=3 (8B) + pad bits
      (int)((tensor_d0_units & 0xFFFFu) << 16),             // td0[15:0]
      (int)((tensor_d0_units >> 16) |
            ((tensor_d1 & 0xFFFFu) << 16)),                 // td0[31:16]|td1[15:0]
      (int)((tensor_d1 >> 16) | (row_units << 16)),         // td1[31:16]|tile_dim0
      (int)rows,                                            // tile_dim1 (tile_dim2=0)
      (int)stride_units,                                    // dim0_stride[31:0]
      0, 0 };
  tdm_g2_t gz4 = {0, 0, 0, 0};
  tdm_g4_t gz8 = {0, 0, 0, 0, 0, 0, 0, 0};
  __builtin_amdgcn_tensor_load_to_lds(g0, g1, gz4, gz4, gz8, 0);
}
#endif

// ------------------------------ fp32 -> f16 --------------------------------
__global__ __launch_bounds__(256) void mha_cvt_f16(
    const float* __restrict__ src, half_t* __restrict__ dst, int n4) {
  int i = blockIdx.x * 256 + threadIdx.x;
  if (i < n4) {
    float4 f = ((const float4*)src)[i];
    union { half_t h[4]; uint2 u; } pk;
    pk.h[0] = (half_t)f.x; pk.h[1] = (half_t)f.y;
    pk.h[2] = (half_t)f.z; pk.h[3] = (half_t)f.w;
    ((uint2*)dst)[i] = pk.u;
  }
}

// ----------------------- tiled WMMA GEMM (f16 x f16 + bias) ----------------
// C[M,N] = A[M,K] @ B[K,N] + bias ; block tile 128x128, 8 waves of 32x64
__global__ __launch_bounds__(256) void mha_gemm_wmma(
    const half_t* __restrict__ A, const half_t* __restrict__ B,
    const float* __restrict__ bias,
    half_t* __restrict__ Ch, float* __restrict__ Cf,
    int M, int N, int K) {
  __shared__ half_t sA[128][32];
  __shared__ half_t sB[32][136];   // +8 halfs pad (TDM: pad 4 DWORDs / 64 DWORDs)

  const int tid  = threadIdx.x;
  const int lane = tid & 31;
  const int wave = tid >> 5;
  const int fr   = lane & 15;
  const int hi   = lane >> 4;
  const int bm   = blockIdx.y * 128;
  const int bn   = blockIdx.x * 128;
  const int wm   = (wave & 3) * 32;
  const int wn   = (wave >> 2) * 64;

  const v8f zero8 = {0.f,0.f,0.f,0.f,0.f,0.f,0.f,0.f};
  v8f acc[2][4];
#pragma unroll
  for (int i = 0; i < 2; ++i)
#pragma unroll
    for (int j = 0; j < 4; ++j) acc[i][j] = zero8;

#if USE_TDM
  const unsigned ldsA = lds_offset_of(&sA[0][0]);
  const unsigned ldsB = lds_offset_of(&sB[0][0]);
  // B tile rows are 256B = 64 DWORDs -> pad_interval enc 5; pad 4 DWORDs -> enc 3
  const unsigned padB = (1u << 20) | (5u << 22) | (3u << 25);
#endif

  for (int k0 = 0; k0 < K; k0 += 32) {
    __syncthreads();
#if USE_TDM
    if (wave == 0) {
      // A tile: 128 rows x 64B ; tensor row = K halfs = K/4 units of 8B
      tdm_load_2d(&A[(size_t)bm * K + k0], ldsA,
                  /*row_units=*/8, /*rows=*/128, /*stride=*/(unsigned)K / 4,
                  (unsigned)K / 4, (unsigned)M, 0u);
      // B tile: 32 rows x 256B ; tensor row = N halfs = N/4 units of 8B
      tdm_load_2d(&B[(size_t)k0 * N + bn], ldsB,
                  /*row_units=*/32, /*rows=*/32, /*stride=*/(unsigned)N / 4,
                  (unsigned)N / 4, (unsigned)K, padB);
      __builtin_amdgcn_s_wait_tensorcnt(0);
    }
    __syncthreads();
#else
#pragma unroll
    for (int i = 0; i < 2; ++i) {
      int idx = tid + 256 * i;
      int ar = idx >> 2, ac = (idx & 3) * 8;
      *(uint4*)&sA[ar][ac] =
          *(const uint4*)&A[(size_t)(bm + ar) * K + k0 + ac];
      int br = idx >> 4, bc = (idx & 15) * 8;
      *(uint4*)&sB[br][bc] =
          *(const uint4*)&B[(size_t)(k0 + br) * N + bn + bc];
    }
    __syncthreads();
    if (k0 + 32 < K) {   // global_prefetch_b8 of next tiles
      __builtin_prefetch(&A[(size_t)(bm + (tid >> 1)) * K + k0 + 32], 0, 0);
      __builtin_prefetch(&B[(size_t)(k0 + 32 + (tid >> 3)) * N + bn], 0, 0);
    }
#endif

    frag16 af[2];
#pragma unroll
    for (int mt = 0; mt < 2; ++mt) {
      int row = wm + mt * 16 + fr;
#pragma unroll
      for (int p = 0; p < 8; ++p) {
        int kb = hi * 16 + 2 * p;
        af[mt].u[p] = *(const unsigned*)&sA[row][kb];
      }
    }
    frag16 bf[4];
#pragma unroll
    for (int nt = 0; nt < 4; ++nt) {
      int col = wn + nt * 16 + fr;
#pragma unroll
      for (int p = 0; p < 8; ++p) {
        int kb = hi * 16 + 2 * p;
        unsigned lo = *(const unsigned short*)&sB[kb][col];
        unsigned h2 = *(const unsigned short*)&sB[kb + 1][col];
        bf[nt].u[p] = lo | (h2 << 16);
      }
    }
#pragma unroll
    for (int mt = 0; mt < 2; ++mt)
#pragma unroll
      for (int nt = 0; nt < 4; ++nt)
        acc[mt][nt] = __builtin_amdgcn_wmma_f32_16x16x32_f16(
            false, af[mt].v, false, bf[nt].v, (short)0, acc[mt][nt],
            false, false);
  }

#pragma unroll
  for (int mt = 0; mt < 2; ++mt)
#pragma unroll
    for (int nt = 0; nt < 4; ++nt)
#pragma unroll
      for (int r = 0; r < 8; ++r) {
        int m = bm + wm + mt * 16 + hi * 8 + r;
        int n = bn + wn + nt * 16 + fr;
        float val = acc[mt][nt][r] + bias[n];
        if (Ch) Ch[(size_t)m * N + n] = (half_t)val;
        if (Cf) Cf[(size_t)m * N + n] = val;
      }
}

// --------------------------- scores = qK^T /8 + mask -----------------------
// block = 4 waves; each block handles one (b, h, 16-query tile)
__global__ __launch_bounds__(128) void mha_scores(
    const half_t* __restrict__ qP, const half_t* __restrict__ kP,
    const unsigned char* __restrict__ mask, float* __restrict__ attn) {
  const int tid = threadIdx.x, lane = tid & 31, wave = tid >> 5;
  const int fr = lane & 15, hi = lane >> 4;
  const int qt = blockIdx.x, h = blockIdx.y, b = blockIdx.z;

  const half_t* qbase = qP + ((size_t)b * SS + qt * 16) * DM + h * DK;
  frag16 af[2];
#pragma unroll
  for (int c = 0; c < 2; ++c)
#pragma unroll
    for (int p = 0; p < 8; ++p) {
      int kb = c * 32 + hi * 16 + 2 * p;
      af[c].u[p] = *(const unsigned*)&qbase[(size_t)fr * DM + kb];
    }

  const size_t mrow = (size_t)b * SS * SS + ((size_t)qt * 16) * SS;
  float* arow = attn + (((size_t)(b * NH + h) * SS) + (size_t)qt * 16) * SS;
  const v8f zero8 = {0.f,0.f,0.f,0.f,0.f,0.f,0.f,0.f};

  for (int nt = wave; nt < SS / 16; nt += 4) {
    const half_t* kbase = kP + ((size_t)b * SS + nt * 16) * DM + h * DK;
    frag16 bf[2];
#pragma unroll
    for (int c = 0; c < 2; ++c)
#pragma unroll
      for (int p = 0; p < 8; ++p) {
        int kb = c * 32 + hi * 16 + 2 * p;
        bf[c].u[p] = *(const unsigned*)&kbase[(size_t)fr * DM + kb];
      }
    v8f acc = zero8;
    acc = __builtin_amdgcn_wmma_f32_16x16x32_f16(false, af[0].v, false,
                                                 bf[0].v, (short)0, acc,
                                                 false, false);
    acc = __builtin_amdgcn_wmma_f32_16x16x32_f16(false, af[1].v, false,
                                                 bf[1].v, (short)0, acc,
                                                 false, false);
#pragma unroll
    for (int r = 0; r < 8; ++r) {
      int m = hi * 8 + r;
      int n = nt * 16 + fr;
      float v = acc[r] * 0.125f;
      if (mask[mrow + (size_t)m * SS + n]) v = -1e9f;
      arow[(size_t)m * SS + n] = v;
    }
  }
}

// ------------------------------ row softmax --------------------------------
__global__ __launch_bounds__(256) void mha_softmax(float* __restrict__ attn) {
  __shared__ float red[256];
  float* p = attn + (size_t)blockIdx.x * SS;
  int t = threadIdx.x;
  float vals[8];
  float mx = -3.4e38f;
#pragma unroll
  for (int i = 0; i < 8; ++i) { vals[i] = p[t + i * 256]; mx = fmaxf(mx, vals[i]); }
  red[t] = mx; __syncthreads();
  for (int s = 128; s > 0; s >>= 1) {
    if (t < s) red[t] = fmaxf(red[t], red[t + s]);
    __syncthreads();
  }
  mx = red[0]; __syncthreads();
  float sum = 0.f;
#pragma unroll
  for (int i = 0; i < 8; ++i) { vals[i] = __expf(vals[i] - mx); sum += vals[i]; }
  red[t] = sum; __syncthreads();
  for (int s = 128; s > 0; s >>= 1) {
    if (t < s) red[t] += red[t + s];
    __syncthreads();
  }
  float inv = 1.f / red[0];
#pragma unroll
  for (int i = 0; i < 8; ++i) p[t + i * 256] = vals[i] * inv;
}

// --------------------------- context = attn @ V ----------------------------
// block = 4 waves, each wave owns a 16-query tile; V tile staged in LDS
__global__ __launch_bounds__(128) void mha_context(
    const float* __restrict__ attn, const half_t* __restrict__ vP,
    half_t* __restrict__ ctx) {
  __shared__ half_t sV[32][64];
  const int tid = threadIdx.x, lane = tid & 31, wave = tid >> 5;
  const int fr = lane & 15, hi = lane >> 4;
  const int h = blockIdx.y, b = blockIdx.z;
  const int qt = blockIdx.x * 4 + wave;

  const float* arow =
      attn + (((size_t)(b * NH + h) * SS) + (size_t)qt * 16) * SS;
  const v8f zero8 = {0.f,0.f,0.f,0.f,0.f,0.f,0.f,0.f};
  v8f acc[4];
#pragma unroll
  for (int i = 0; i < 4; ++i) acc[i] = zero8;

#if USE_TDM
  const unsigned ldsV = lds_offset_of(&sV[0][0]);
#endif

  for (int kk = 0; kk < SS / 32; ++kk) {
    __syncthreads();
#if USE_TDM
    if (wave == 0) {
      // V tile: 32 rows x 128B ; tensor row = DM halfs = DM/4 units of 8B
      tdm_load_2d(&vP[((size_t)b * SS + kk * 32) * DM + h * DK], ldsV,
                  /*row_units=*/16, /*rows=*/32, /*stride=*/DM / 4,
                  DM / 4, BB * SS, 0u);
      __builtin_amdgcn_s_wait_tensorcnt(0);
    }
    __syncthreads();
#else
#pragma unroll
    for (int i = 0; i < 2; ++i) {
      int idx = tid + 128 * i;
      int vr = idx >> 3, vc = (idx & 7) * 8;
      *(uint4*)&sV[vr][vc] =
          *(const uint4*)&vP[((size_t)b * SS + kk * 32 + vr) * DM + h * DK + vc];
    }
    __syncthreads();
#endif

    frag16 af;
#pragma unroll
    for (int p = 0; p < 8; ++p) {
      int kb = hi * 16 + 2 * p;
      float2 f = *(const float2*)&arow[(size_t)fr * SS + kk * 32 + kb];
      af.h[2 * p]     = (_Float16)f.x;
      af.h[2 * p + 1] = (_Float16)f.y;
    }
#pragma unroll
    for (int nt = 0; nt < 4; ++nt) {
      frag16 bf;
#pragma unroll
      for (int p = 0; p < 8; ++p) {
        int kb = hi * 16 + 2 * p;
        int col = nt * 16 + fr;
        unsigned lo = *(const unsigned short*)&sV[kb][col];
        unsigned h2 = *(const unsigned short*)&sV[kb + 1][col];
        bf.u[p] = lo | (h2 << 16);
      }
      acc[nt] = __builtin_amdgcn_wmma_f32_16x16x32_f16(
          false, af.v, false, bf.v, (short)0, acc[nt], false, false);
    }
  }
#pragma unroll
  for (int nt = 0; nt < 4; ++nt)
#pragma unroll
    for (int r = 0; r < 8; ++r) {
      int m = qt * 16 + hi * 8 + r;
      int n = h * DK + nt * 16 + fr;
      ctx[((size_t)b * SS + m) * DM + n] = (half_t)acc[nt][r];
    }
}

// ---------------------- residual add + LayerNorm ---------------------------
__global__ __launch_bounds__(256) void mha_add_ln(
    const float* __restrict__ xin, const float* __restrict__ resid,
    const float* __restrict__ g, const float* __restrict__ be,
    float* __restrict__ out) {
  __shared__ float red[256];
  const size_t row = blockIdx.x;
  const float* xr = xin + row * DM;
  const float* rr = resid + row * DM;
  int t = threadIdx.x;
  float x[4];
  float s = 0.f;
#pragma unroll
  for (int i = 0; i < 4; ++i) { x[i] = xr[t + i * 256] + rr[t + i * 256]; s += x[i]; }
  red[t] = s; __syncthreads();
  for (int k = 128; k > 0; k >>= 1) { if (t < k) red[t] += red[t + k]; __syncthreads(); }
  float mean = red[0] * (1.f / DM); __syncthreads();
  float s2 = 0.f;
#pragma unroll
  for (int i = 0; i < 4; ++i) { float d = x[i] - mean; s2 += d * d; }
  red[t] = s2; __syncthreads();
  for (int k = 128; k > 0; k >>= 1) { if (t < k) red[t] += red[t + k]; __syncthreads(); }
  float rstd = rsqrtf(red[0] * (1.f / DM) + 1e-5f);
#pragma unroll
  for (int i = 0; i < 4; ++i) {
    int c = t + i * 256;
    out[row * DM + c] = (x[i] - mean) * rstd * g[c] + be[c];
  }
}

// ---------------------------------------------------------------------------
extern "C" void kernel_launch(void* const* d_in, const int* in_sizes, int n_in,
                              void* d_out, int out_size, void* d_ws, size_t ws_size,
                              hipStream_t stream) {
  const float* Q  = (const float*)d_in[0];
  const float* K  = (const float*)d_in[1];
  const float* V  = (const float*)d_in[2];
  const unsigned char* mask = (const unsigned char*)d_in[3];
  const float* Wq = (const float*)d_in[4];
  const float* bq = (const float*)d_in[5];
  const float* Wk = (const float*)d_in[6];
  const float* bk = (const float*)d_in[7];
  const float* Wv = (const float*)d_in[8];
  const float* bv = (const float*)d_in[9];
  const float* Wo = (const float*)d_in[10];
  const float* bo = (const float*)d_in[11];
  const float* g  = (const float*)d_in[12];
  const float* be = (const float*)d_in[13];

  float* out  = (float*)d_out;
  float* attn = out + (size_t)BB * SS * DM;   // [B,H,S,S]

  // workspace layout (all 256B aligned)
  char* w = (char*)d_ws;
  size_t off = 0;
  auto alloc = [&](size_t bytes) {
    void* p = w + off;
    off += (bytes + 255) & ~(size_t)255;
    return p;
  };
  const size_t actH = (size_t)BB * SS * DM * sizeof(half_t);  // 16 MB
  const size_t wH   = (size_t)DM * DM * sizeof(half_t);       // 2 MB
  half_t* QH  = (half_t*)alloc(actH);
  half_t* KH  = (half_t*)alloc(actH);
  half_t* VH  = (half_t*)alloc(actH);
  half_t* WqH = (half_t*)alloc(wH);
  half_t* WkH = (half_t*)alloc(wH);
  half_t* WvH = (half_t*)alloc(wH);
  half_t* WoH = (half_t*)alloc(wH);
  half_t* qP  = (half_t*)alloc(actH);
  half_t* kP  = (half_t*)alloc(actH);
  half_t* vP  = (half_t*)alloc(actH);
  half_t* ctx = QH;               // QH free after q projection
  float*  otmp = (float*)KH;      // KH..VH region (32 MB) free after context

  const int nAct = BB * SS * DM;  // 8388608
  const int nW   = DM * DM;       // 1048576

  // 1) fp32 -> f16 conversions
  mha_cvt_f16<<<(nAct / 4 + 255) / 256, 256, 0, stream>>>(Q, QH, nAct / 4);
  mha_cvt_f16<<<(nAct / 4 + 255) / 256, 256, 0, stream>>>(K, KH, nAct / 4);
  mha_cvt_f16<<<(nAct / 4 + 255) / 256, 256, 0, stream>>>(V, VH, nAct / 4);
  mha_cvt_f16<<<(nW / 4 + 255) / 256, 256, 0, stream>>>(Wq, WqH, nW / 4);
  mha_cvt_f16<<<(nW / 4 + 255) / 256, 256, 0, stream>>>(Wk, WkH, nW / 4);
  mha_cvt_f16<<<(nW / 4 + 255) / 256, 256, 0, stream>>>(Wv, WvH, nW / 4);
  mha_cvt_f16<<<(nW / 4 + 255) / 256, 256, 0, stream>>>(Wo, WoH, nW / 4);

  // 2) QKV projections (M=8192, N=1024, K=1024)
  dim3 ggrid(DM / 128, (BB * SS) / 128);
  mha_gemm_wmma<<<ggrid, 256, 0, stream>>>(QH, WqH, bq, qP, nullptr,
                                           BB * SS, DM, DM);
  mha_gemm_wmma<<<ggrid, 256, 0, stream>>>(KH, WkH, bk, kP, nullptr,
                                           BB * SS, DM, DM);
  mha_gemm_wmma<<<ggrid, 256, 0, stream>>>(VH, WvH, bv, vP, nullptr,
                                           BB * SS, DM, DM);

  // 3) scores + mask -> attn region of d_out (raw, pre-softmax)
  mha_scores<<<dim3(SS / 16, NH, BB), 128, 0, stream>>>(qP, kP, mask, attn);

  // 4) softmax in place
  mha_softmax<<<BB * NH * SS, 256, 0, stream>>>(attn);

  // 5) context = attn @ V  (f16 out, [B,S,H*Dv] layout)
  mha_context<<<dim3(SS / 64, NH, BB), 128, 0, stream>>>(attn, vP, ctx);

  // 6) output projection (f32 out)
  mha_gemm_wmma<<<ggrid, 256, 0, stream>>>(ctx, WoH, bo, nullptr, otmp,
                                           BB * SS, DM, DM);

  // 7) residual + LayerNorm -> d_out
  mha_add_ln<<<BB * SS, 256, 0, stream>>>(otmp, Q, g, be, out);
}
